// Cross_Attention_61873298866510
// MI455X (gfx1250) — matrix operand order repaired and checked by reading, not compiled, not run on hardware
//
#include <hip/hip_runtime.h>
#include <hip/hip_bf16.h>

// B=4, C=128, Hs=Ws=64 -> N=4096 spatial, NUM_HEADS=4, HEAD_DIM=32.
//   proj_kernel: Qh[b,h,n,32] = scale*(Wq@query), Kh[b,h,n,32] = Wk@key,
//                Vt[b,h,32,n] = (Wv@key) transposed.  All f16, via WMMA.
//   attn_kernel: flash attention; block = 4 waves = one (b,h), 64 query rows.
//                K/V key-blocks (32 keys) staged block-wide into LDS with
//                GLOBAL_LOAD_ASYNC_TO_LDS_B128 (ASYNCcnt double-buffer),
//                v_wmma_f32_16x16x32_f16 for S and P@V, exp2-domain softmax.

typedef __attribute__((ext_vector_type(16))) _Float16 v16h;
typedef __attribute__((ext_vector_type(8)))  _Float16 v8h;
typedef __attribute__((ext_vector_type(8)))  float    v8f;
typedef __attribute__((ext_vector_type(4)))  float    v4f;
typedef __attribute__((ext_vector_type(4)))  int      v4i;

#if defined(__AMDGCN__) && __has_builtin(__builtin_amdgcn_exp2f)
#define EXP2F(x) __builtin_amdgcn_exp2f(x)
#else
#define EXP2F(x) __builtin_exp2f(x)
#endif
#if defined(__AMDGCN__) && __has_builtin(__builtin_amdgcn_rcpf)
#define RCPF(x) __builtin_amdgcn_rcpf(x)
#else
#define RCPF(x) (1.0f / (x))
#endif

#if defined(__AMDGCN__) && __has_builtin(__builtin_amdgcn_global_load_async_to_lds_b128)
#define HAVE_ASYNC_LDS 1
typedef __attribute__((address_space(1))) v4i* g_v4i_ptr;
typedef __attribute__((address_space(3))) v4i* l_v4i_ptr;
#else
#define HAVE_ASYNC_LDS 0
#endif

// 16-byte global -> LDS copy (async on gfx1250 when the builtin exists).
static __device__ __forceinline__ void copy16B(const _Float16* g, _Float16* l) {
#if HAVE_ASYNC_LDS
  __builtin_amdgcn_global_load_async_to_lds_b128((g_v4i_ptr)g, (l_v4i_ptr)l, 0, 0);
#else
  *(v8h*)l = *(const v8h*)g;
#endif
}
static __device__ __forceinline__ void wait_async_keep_one_stage() {
#if HAVE_ASYNC_LDS
  asm volatile("s_wait_asynccnt 0x2" ::: "memory");  // 2 copies per stage
#endif
}
static __device__ __forceinline__ void wait_async_all() {
#if HAVE_ASYNC_LDS
  asm volatile("s_wait_asynccnt 0x0" ::: "memory");
#endif
}

static __device__ __forceinline__ v16h cat16(v8h lo, v8h hi) {
  v16h r;
#pragma unroll
  for (int i = 0; i < 8; ++i) { r[i] = lo[i]; r[i + 8] = hi[i]; }
  return r;
}

static __device__ __forceinline__ v8f wmma_f16(v16h a, v16h b, v8f c) {
  // (neg_a, A, neg_b, B, c_mod, C, reuse_a, reuse_b)
  return __builtin_amdgcn_wmma_f32_16x16x32_f16(false, a, false, b, (short)0, c,
                                                false, false);
}

// ---------------------------------------------------------------------------
// Projection: out[o,n] = sum_c W[o,c] * X[b,c,n], written as f16 head layouts.
// Block = 256 threads (8 waves), handles (b, 64-column spatial tile).
// ---------------------------------------------------------------------------
__global__ __launch_bounds__(256) void proj_kernel(
    const float* __restrict__ query, const float* __restrict__ key,
    const float* __restrict__ Wq, const float* __restrict__ Wk,
    const float* __restrict__ Wv, _Float16* __restrict__ Qh,
    _Float16* __restrict__ Kh, _Float16* __restrict__ Vt) {
  __shared__ __align__(16) _Float16 xT[64 * 136];  // [n][c], padded row 136
  const int b    = blockIdx.y;
  const int n0   = blockIdx.x * 64;
  const int tid  = threadIdx.x;
  const int lane = tid & 31;
  const int wv   = tid >> 5;
  const int cl   = lane & 15;
  const int hi   = lane >> 4;
  const int dlt  = hi * 8;  // A-fragment K-offset per lane half

  for (int phase = 0; phase < 2; ++phase) {
    const float* src = phase ? key : query;
    __syncthreads();
#pragma unroll 4
    for (int idx = tid; idx < 128 * 64; idx += 256) {
      const int c = idx >> 6, n = idx & 63;
      xT[n * 136 + c] = (_Float16)src[((size_t)b * 128 + c) * 4096 + n0 + n];
    }
    __syncthreads();
    const int nW = phase ? 2 : 1;
    for (int wi = 0; wi < nW; ++wi) {
      const float* W    = phase ? (wi ? Wv : Wk) : Wq;
      // Fold softmax scale (1/sqrt(32)) and log2(e) into Wq.
      const float scale = phase ? 1.0f : 0.25507904f;
      for (int t = wv; t < 32; t += 8) {  // 32 output 16x16 tiles, 4 per wave
        const int ot = t >> 2, nt = t & 3;
        v8f acc = {};
#pragma unroll
        for (int kk = 0; kk < 4; ++kk) {  // K = 128 in 4 steps of 32
          // A fragment (W rows): split-8 K pattern.
          const float* wr = W + (size_t)(ot * 16 + cl) * 128 + kk * 32 + dlt;
          v16h a;
#pragma unroll
          for (int i = 0; i < 8; ++i) {
            a[i]     = (_Float16)(wr[i] * scale);
            a[i + 8] = (_Float16)(wr[i + 16] * scale);
          }
          // B fragment (X cols): split-8 K pattern.
          const _Float16* xr = xT + (nt * 16 + cl) * 136 + kk * 32 + dlt;
          const v16h bf = cat16(*(const v8h*)xr, *(const v8h*)(xr + 16));
          acc = wmma_f16(a, bf, acc);
        }
        // D fragment: N = lane&15 (spatial), M = r + 8*hi (out channel).
        const int    Ncol  = n0 + nt * 16 + cl;
        const int    h     = ot >> 1;
        const size_t bh    = (size_t)b * 4 + h;
        const int    dbase = (ot & 1) * 16 + hi * 8;  // head-dim index of acc[0]
        if (phase == 0) {
          v8h hv;
#pragma unroll
          for (int r = 0; r < 8; ++r) hv[r] = (_Float16)acc[r];
          *(v8h*)(Qh + (bh * 4096 + Ncol) * 32 + dbase) = hv;
        } else if (wi == 0) {
          v8h hv;
#pragma unroll
          for (int r = 0; r < 8; ++r) hv[r] = (_Float16)acc[r];
          *(v8h*)(Kh + (bh * 4096 + Ncol) * 32 + dbase) = hv;
        } else {  // V transposed: Vt[bh][d][n]
          _Float16* dst = Vt + bh * 131072 + (size_t)dbase * 4096 + Ncol;
#pragma unroll
          for (int r = 0; r < 8; ++r) dst[(size_t)r * 4096] = (_Float16)acc[r];
        }
      }
    }
  }
}

// ---------------------------------------------------------------------------
// Flash attention: block = 4 waves (128 thr) on one (b,h); wave = 16 q rows.
// K/V staged into LDS block-wide, double-buffered via ASYNCcnt.
// ---------------------------------------------------------------------------
__global__ __launch_bounds__(128) void attn_kernel(
    const _Float16* __restrict__ Qh, const _Float16* __restrict__ Kh,
    const _Float16* __restrict__ Vt, float* __restrict__ out) {
  __shared__ __align__(16) _Float16 kBuf[2][32 * 40];  // [key][d], padded 40
  __shared__ __align__(16) _Float16 vBuf[2][32 * 40];  // [d][key], padded 40
  __shared__ __align__(16) _Float16 pT[4 * 16 * 40];   // per-wave P transpose
  const int tid  = threadIdx.x;
  const int lane = tid & 31;
  const int wv   = tid >> 5;
  const int cl   = lane & 15;
  const int hi   = lane >> 4;
  const int dlt  = hi * 8;
  const int bh   = blockIdx.y;
  const int q0   = blockIdx.x * 64 + wv * 16;

  const _Float16* Qb = Qh + (size_t)bh * 4096 * 32;
  const _Float16* Kb = Kh + (size_t)bh * 4096 * 32;
  const _Float16* Vb = Vt + (size_t)bh * 131072;
  _Float16*       pW = pT + wv * 640;

  // Block-wide copy mapping: thread -> (row = tid>>2, 8-half chunk = tid&3).
  const int crow = tid >> 2;
  const int cch  = (tid & 3) * 8;
  const _Float16* Kg = Kb + (size_t)crow * 32 + cch;    // + stage*32*32
  const _Float16* Vg = Vb + (size_t)crow * 4096 + cch;  // + stage*32
  const int       lofs = crow * 40 + cch;

  // Q A-fragment (held for whole kernel): row = q0+cl, K split-8 pattern.
  const _Float16* qr = Qb + (size_t)(q0 + cl) * 32 + dlt;
  const v16h qf = cat16(*(const v8h*)qr, *(const v8h*)(qr + 16));

  v8f o0 = {}, o1 = {};
  float m[8], l[8];
#pragma unroll
  for (int r = 0; r < 8; ++r) { m[r] = -3.0e38f; l[r] = 0.0f; }
  const v8f zero = {};

  // Prologue: stage 0 into buffer 0.
  copy16B(Kg, &kBuf[0][lofs]);
  copy16B(Vg, &vBuf[0][lofs]);

  for (int jb = 0; jb < 128; ++jb) {
    const int bi = jb & 1;
    if (jb + 1 < 128) {
      // Prefetch next stage into the other buffer (its readers finished at
      // the barrier that ended iteration jb-1).
      copy16B(Kg + (size_t)(jb + 1) * 1024, &kBuf[bi ^ 1][lofs]);
      copy16B(Vg + (size_t)(jb + 1) * 32, &vBuf[bi ^ 1][lofs]);
      wait_async_keep_one_stage();  // my stage-jb copies complete
    } else {
      wait_async_all();
    }
    __syncthreads();  // everyone's stage-jb copies visible

    // K B-fragments: col = key (cl / 16+cl), halves = d (split-8 K pattern)
    const _Float16* kr0 = &kBuf[bi][cl * 40 + dlt];
    const _Float16* kr1 = kr0 + 16 * 40;
    const v16h kf0 = cat16(*(const v8h*)kr0, *(const v8h*)(kr0 + 16));
    const v16h kf1 = cat16(*(const v8h*)kr1, *(const v8h*)(kr1 + 16));
    // V B-fragments: col = d (cl / 16+cl), halves = key (split-8 K pattern)
    const _Float16* vr0 = &vBuf[bi][cl * 40 + dlt];
    const _Float16* vr1 = vr0 + 16 * 40;
    const v16h vf0 = cat16(*(const v8h*)vr0, *(const v8h*)(vr0 + 16));
    const v16h vf1 = cat16(*(const v8h*)vr1, *(const v8h*)(vr1 + 16));

    v8f s0 = wmma_f16(qf, kf0, zero);  // scores already in exp2 domain
    v8f s1 = wmma_f16(qf, kf1, zero);

    // Online softmax.  C layout: row M = r + 8*hi, col = lane&15 -> row stats
    // reduce across the 16-lane groups (masks 1,2,4,8 stay inside the group).
#pragma unroll
    for (int r = 0; r < 8; ++r) {
      float c = fmaxf(s0[r], s1[r]);
      c = fmaxf(c, __shfl_xor(c, 1, 32));
      c = fmaxf(c, __shfl_xor(c, 2, 32));
      c = fmaxf(c, __shfl_xor(c, 4, 32));
      c = fmaxf(c, __shfl_xor(c, 8, 32));
      const float mn    = fmaxf(m[r], c);
      const float alpha = EXP2F(m[r] - mn);
      m[r] = mn;
      const float p0 = EXP2F(s0[r] - mn);
      const float p1 = EXP2F(s1[r] - mn);
      float rs = p0 + p1;
      rs += __shfl_xor(rs, 1, 32);
      rs += __shfl_xor(rs, 2, 32);
      rs += __shfl_xor(rs, 4, 32);
      rs += __shfl_xor(rs, 8, 32);
      l[r]  = l[r] * alpha + rs;
      o0[r] *= alpha;
      o1[r] *= alpha;
      // Stage P into LDS [M][K] for A-fragment pickup.
      const int M = r + hi * 8;
      pW[M * 40 + cl]      = (_Float16)p0;
      pW[M * 40 + cl + 16] = (_Float16)p1;
    }
    // Per-wave DS ops are in-order; wait then read back in A layout.
    asm volatile("s_wait_dscnt 0" ::: "memory");
    const _Float16* pr = pW + cl * 40 + dlt;
    const v16h pf = cat16(*(const v8h*)pr, *(const v8h*)(pr + 16));
    o0 = wmma_f16(pf, vf0, o0);
    o1 = wmma_f16(pf, vf1, o1);

    __syncthreads();  // all reads of buf[bi] done before it is re-filled
  }

#pragma unroll
  for (int r = 0; r < 8; ++r) {
    const float inv = RCPF(l[r]);
    o0[r] *= inv;
    o1[r] *= inv;
  }
  // O fragment: M = query offset (r + 8*hi), N = head-dim (cl / 16+cl).
  // out[b][h*32+d][q] with q contiguous over r -> two b128 stores per fragment.
  const int b = bh >> 2, h = bh & 3;
  float* ob0 = out + (size_t)(b * 128 + h * 32 + cl) * 4096 + q0 + hi * 8;
  v4f t;
  t[0] = o0[0]; t[1] = o0[1]; t[2] = o0[2]; t[3] = o0[3];
  *(v4f*)ob0 = t;
  t[0] = o0[4]; t[1] = o0[5]; t[2] = o0[6]; t[3] = o0[7];
  *(v4f*)(ob0 + 4) = t;
  float* ob1 = ob0 + (size_t)16 * 4096;
  t[0] = o1[0]; t[1] = o1[1]; t[2] = o1[2]; t[3] = o1[3];
  *(v4f*)ob1 = t;
  t[0] = o1[4]; t[1] = o1[5]; t[2] = o1[6]; t[3] = o1[7];
  *(v4f*)(ob1 + 4) = t;
}

// ---------------------------------------------------------------------------
extern "C" void kernel_launch(void* const* d_in, const int* in_sizes, int n_in,
                              void* d_out, int out_size, void* d_ws,
                              size_t ws_size, hipStream_t stream) {
  (void)in_sizes; (void)n_in; (void)out_size; (void)ws_size;
  const float* query = (const float*)d_in[0];
  const float* key   = (const float*)d_in[1];
  const float* Wq    = (const float*)d_in[2];
  const float* Wk    = (const float*)d_in[3];
  const float* Wv    = (const float*)d_in[4];

  _Float16* Qh = (_Float16*)d_ws;            // 16 * 4096 * 32 halves = 4 MB
  _Float16* Kh = Qh + (size_t)16 * 4096 * 32;
  _Float16* Vt = Kh + (size_t)16 * 4096 * 32;

  proj_kernel<<<dim3(64, 4), 256, 0, stream>>>(query, key, Wq, Wk, Wv, Qh, Kh, Vt);
  attn_kernel<<<dim3(64, 16), 128, 0, stream>>>(Qh, Kh, Vt, (float*)d_out);
}
